// SingleHeadSelfAttention_31911607009679
// MI455X (gfx1250) — matrix operand order repaired and checked by reading, not compiled, not run on hardware
//
#include <hip/hip_runtime.h>
#include <hip/hip_bf16.h>

#define D_MODEL 256
#define SEQ_T   4096
#define N_BATCH 4

typedef __attribute__((ext_vector_type(16))) __bf16 v16bf;
typedef __attribute__((ext_vector_type(8)))  __bf16 v8bf;
typedef __attribute__((ext_vector_type(8)))  float  v8f;

// ---- WMMA wrapper: D = A(16x32 bf16) * B(32x16 bf16) + C(16x16 f32) ----
__device__ __forceinline__ v8f wmma_bf16(v16bf a, v16bf b, v8f c) {
  return __builtin_amdgcn_wmma_f32_16x16x32_bf16(false, a, false, b, (short)0, c,
                                                 false, false);
}

// A-fragment (16-bit, 16x32, MxK). Lane (h = lane>>4, m = lane&15) holds row m,
// K = kc+16h .. kc+16h+15 (contiguous).  `row` = row-major row pointer.
__device__ __forceinline__ v16bf frag_a_bf(const __bf16* __restrict__ row, int kc, int h) {
  v8bf lo = *(const v8bf*)(row + kc + 16 * h);
  v8bf hi = *(const v8bf*)(row + kc + 16 * h + 8);
  v16bf r;
#pragma unroll
  for (int i = 0; i < 8; ++i) { r[i] = lo[i]; r[i + 8] = hi[i]; }
  return r;
}

// B-fragment (16-bit, 32x16, KxN). Lane (h, n) holds column n,
// K = kc+16h .. kc+16h+15 (contiguous). `col` = pointer to that column's
// K-values stored contiguously (row of the transposed matrix).
__device__ __forceinline__ v16bf frag_b_bf(const __bf16* __restrict__ col, int kc, int h) {
  v8bf lo = *(const v8bf*)(col + kc + 16 * h);
  v8bf hi = *(const v8bf*)(col + kc + 16 * h + 8);
  v16bf r;
#pragma unroll
  for (int i = 0; i < 8; ++i) { r[i] = lo[i]; r[i + 8] = hi[i]; }
  return r;
}

// ============ Kernel 0: transpose + convert weights to bf16 ============
// W (rows x cols, f32, row-major)  ->  Wt (cols x rows, bf16, row-major)
// Wt[n][k] = W[k][n]; makes every B-fragment two contiguous 16B loads.
__global__ void wtrans_kernel(const float* __restrict__ W, __bf16* __restrict__ Wt,
                              int rows, int cols) {
  int idx = blockIdx.x * blockDim.x + threadIdx.x;
  if (idx >= rows * cols) return;
  int n = idx / rows;      // output row  (original column)
  int k = idx - n * rows;  // output col  (original row) — contiguous writes
  Wt[(size_t)n * rows + k] = (__bf16)W[(size_t)k * cols + n];
}

// ===================== Kernel 1: QKV projection =====================
// qkv = x @ W_qkv + b_qkv  (x f32 -> bf16 fragments, weights pre-transposed
// bf16). Writes Q,K row-major bf16 and V transposed Vt[b][feat][t] bf16.
__global__ void qkv_kernel(const float* __restrict__ x, const __bf16* __restrict__ Wt,
                           const float* __restrict__ bqkv,
                           __bf16* __restrict__ Qb, __bf16* __restrict__ Kb,
                           __bf16* __restrict__ Vt) {
  const int lane = threadIdx.x & 31;
  const int wave = threadIdx.x >> 5;
  const int h = lane >> 4;
  const int n = lane & 15;
  const int row0 = blockIdx.x * 64 + wave * 16;  // flat token tile base (B*T space)

  // Preload A fragments: 16 tokens x 256 features, 8 chunks of K=32.
  const float* xrow = x + (size_t)(row0 + n) * D_MODEL;
  v16bf afrag[8];
#pragma unroll
  for (int c = 0; c < 8; ++c) {
#pragma unroll
    for (int i = 0; i < 8; ++i) {
      afrag[c][i]     = (__bf16)xrow[c * 32 + 16 * h + i];
      afrag[c][i + 8] = (__bf16)xrow[c * 32 + 16 * h + 8 + i];
    }
  }

  const int batch = row0 / SEQ_T;
  const int t0 = row0 - batch * SEQ_T;

  for (int ct = 0; ct < 48; ++ct) {   // 768 output columns / 16
    const int col = ct * 16 + n;
    const __bf16* wcol = Wt + (size_t)col * D_MODEL;  // column's K-values, contiguous
    const float bias = bqkv[col];
    v8f acc;
#pragma unroll
    for (int r = 0; r < 8; ++r) acc[r] = bias;
#pragma unroll
    for (int c = 0; c < 8; ++c)
      acc = wmma_bf16(afrag[c], frag_b_bf(wcol, c * 32, h), acc);

    if (ct < 16) {          // Q, row-major
#pragma unroll
      for (int r = 0; r < 8; ++r)
        Qb[(size_t)(row0 + r + 8 * h) * D_MODEL + col] = (__bf16)acc[r];
    } else if (ct < 32) {   // K, row-major
#pragma unroll
      for (int r = 0; r < 8; ++r)
        Kb[(size_t)(row0 + r + 8 * h) * D_MODEL + (col - D_MODEL)] = (__bf16)acc[r];
    } else {                // V transposed: Vt[b][feat][t]; token contiguous
      const int feat = col - 2 * D_MODEL;
      v8bf pv;
#pragma unroll
      for (int r = 0; r < 8; ++r) pv[r] = (__bf16)acc[r];
      *(v8bf*)(Vt + ((size_t)batch * D_MODEL + feat) * SEQ_T + t0 + 8 * h) = pv;
    }
  }
}

// ===================== Kernel 2: causal flash attention =====================
// Per wave: one 16-query tile. Computes S^T = K*Q^T (D rows = keys, cols =
// queries) so softmax rows live per-lane (+ one xor-16 shuffle), then
// ctx^T += V^T * P^T with 16 f32 accumulator tiles covering d=256.
__global__ void attn_kernel(const __bf16* __restrict__ Qb, const __bf16* __restrict__ Kb,
                            const __bf16* __restrict__ Vt, __bf16* __restrict__ ctxb) {
  const int lane = threadIdx.x & 31;
  const int wave = threadIdx.x >> 5;
  const int h = lane >> 4;
  const int n = lane & 15;
  const int b = blockIdx.y;
  const int q0 = blockIdx.x * 64 + wave * 16;  // in-batch query tile base

  const __bf16* Q = Qb + (size_t)b * SEQ_T * D_MODEL;
  const __bf16* K = Kb + (size_t)b * SEQ_T * D_MODEL;
  const __bf16* V = Vt + (size_t)b * D_MODEL * SEQ_T;

  // Q^T B-fragments: lane (h,n) = query q0+n, features contiguous.
  const __bf16* qrow = Q + (size_t)(q0 + n) * D_MODEL;
  v16bf qfrag[8];
#pragma unroll
  for (int c = 0; c < 8; ++c) qfrag[c] = frag_b_bf(qrow, c * 32, h);

  v8f acc[16];
#pragma unroll
  for (int t = 0; t < 16; ++t)
#pragma unroll
    for (int r = 0; r < 8; ++r) acc[t][r] = 0.0f;

  float mrow = -1e30f, lrow = 0.0f;
  const float Cc = 0.0625f * 1.44269504088896f;  // (1/sqrt(d)) * log2(e)
  const int kend = q0 + 15;

  for (int kb = 0; kb <= kend; kb += 32) {
    // ---- S^T tiles: st0 = keys kb..kb+15, st1 = keys kb+16..kb+31 ----
    v8f st0, st1;
#pragma unroll
    for (int r = 0; r < 8; ++r) { st0[r] = 0.0f; st1[r] = 0.0f; }
    const __bf16* k0 = K + (size_t)(kb + n) * D_MODEL;
    const __bf16* k1 = K + (size_t)(kb + 16 + n) * D_MODEL;
#pragma unroll
    for (int c = 0; c < 8; ++c) {
      v16bf a0 = frag_a_bf(k0, c * 32, h);
      v16bf a1 = frag_a_bf(k1, c * 32, h);
      st0 = wmma_bf16(a0, qfrag[c], st0);
      st1 = wmma_bf16(a1, qfrag[c], st1);
    }
    // ---- causal mask (key > query -> -inf) ----
    if (kb + 31 > q0) {
      const int q = q0 + n;
#pragma unroll
      for (int r = 0; r < 8; ++r) {
        st0[r] = (kb + r + 8 * h > q) ? -1e30f : st0[r];
        st1[r] = (kb + 16 + r + 8 * h > q) ? -1e30f : st1[r];
      }
    }
    // ---- online softmax (per-query column = per lane + partner half) ----
    float mx = -1e30f;
#pragma unroll
    for (int r = 0; r < 8; ++r) { mx = fmaxf(mx, st0[r]); mx = fmaxf(mx, st1[r]); }
    mx = fmaxf(mx, __shfl_xor(mx, 16, 32));
    const float mnew = fmaxf(mrow, mx);
    const float cf = exp2f((mrow - mnew) * Cc);
    float p0[8], p1[8], sum = 0.0f;
#pragma unroll
    for (int r = 0; r < 8; ++r) {
      p0[r] = exp2f((st0[r] - mnew) * Cc);
      p1[r] = exp2f((st1[r] - mnew) * Cc);
      sum += p0[r] + p1[r];
    }
    sum += __shfl_xor(sum, 16, 32);
    lrow = lrow * cf + sum;
    mrow = mnew;
#pragma unroll
    for (int t = 0; t < 16; ++t)
#pragma unroll
      for (int r = 0; r < 8; ++r) acc[t][r] *= cf;

    // ---- build P^T B-fragment: one xor-16 exchange of 8 packed bf16 ----
    union PU { v8bf v; unsigned int u[4]; };
    PU own, snd, rcv;
#pragma unroll
    for (int r = 0; r < 8; ++r) {
      own.v[r] = (__bf16)(h ? p1[r] : p0[r]);  // this half's own tile
      snd.v[r] = (__bf16)(h ? p0[r] : p1[r]);  // what the partner half needs
    }
#pragma unroll
    for (int i = 0; i < 4; ++i) rcv.u[i] = __shfl_xor(snd.u[i], 16, 32);
    v16bf pfrag;
#pragma unroll
    for (int j = 0; j < 8; ++j) {
      pfrag[j]     = h ? rcv.v[j] : own.v[j];  // K = 16h + j      (keys j)
      pfrag[j + 8] = h ? own.v[j] : rcv.v[j];  // K = 16h + j + 8  (keys j+8)
    }
    // ---- ctx^T += V^T * P^T over all 16 feature tiles ----
#pragma unroll
    for (int ft = 0; ft < 16; ++ft) {
      const __bf16* vrow = V + (size_t)(ft * 16 + n) * SEQ_T;  // feature row
      v16bf av = frag_a_bf(vrow, kb, h);
      acc[ft] = wmma_bf16(av, pfrag, acc[ft]);
    }
  }

  // ---- epilogue: divide by l, store ctx row-major bf16 ----
  const float inv = 1.0f / lrow;
  __bf16* crow = ctxb + ((size_t)b * SEQ_T + q0 + n) * D_MODEL;
#pragma unroll
  for (int ft = 0; ft < 16; ++ft) {
    v8bf o;
#pragma unroll
    for (int r = 0; r < 8; ++r) o[r] = (__bf16)(acc[ft][r] * inv);
    *(v8bf*)(crow + ft * 16 + 8 * h) = o;
  }
}

// ===================== Kernel 3: output projection =====================
// out = ctx @ W_proj + b_proj  (bf16 A, pre-transposed bf16 weights, f32 out)
__global__ void proj_kernel(const __bf16* __restrict__ ctxb, const __bf16* __restrict__ Wt,
                            const float* __restrict__ bp, float* __restrict__ out) {
  const int lane = threadIdx.x & 31;
  const int wave = threadIdx.x >> 5;
  const int h = lane >> 4;
  const int n = lane & 15;
  const int row0 = blockIdx.x * 64 + wave * 16;

  const __bf16* arow = ctxb + (size_t)(row0 + n) * D_MODEL;
  v16bf afrag[8];
#pragma unroll
  for (int c = 0; c < 8; ++c) afrag[c] = frag_a_bf(arow, c * 32, h);

  for (int ct = 0; ct < 16; ++ct) {
    const int col = ct * 16 + n;
    const __bf16* wcol = Wt + (size_t)col * D_MODEL;
    const float bias = bp[col];
    v8f acc;
#pragma unroll
    for (int r = 0; r < 8; ++r) acc[r] = bias;
#pragma unroll
    for (int c = 0; c < 8; ++c)
      acc = wmma_bf16(afrag[c], frag_b_bf(wcol, c * 32, h), acc);
#pragma unroll
    for (int r = 0; r < 8; ++r)
      out[(size_t)(row0 + r + 8 * h) * D_MODEL + col] = acc[r];
  }
}

// ===================== launch =====================
extern "C" void kernel_launch(void* const* d_in, const int* in_sizes, int n_in,
                              void* d_out, int out_size, void* d_ws, size_t ws_size,
                              hipStream_t stream) {
  (void)in_sizes; (void)n_in; (void)out_size; (void)ws_size;
  const float* x    = (const float*)d_in[0];
  const float* Wqkv = (const float*)d_in[1];
  const float* bqkv = (const float*)d_in[2];
  const float* Wp   = (const float*)d_in[3];
  const float* bp   = (const float*)d_in[4];
  float* out = (float*)d_out;

  const size_t BT = (size_t)N_BATCH * SEQ_T;
  __bf16* Qb    = (__bf16*)d_ws;              // [B*T, 256]
  __bf16* Kb    = Qb + BT * D_MODEL;          // [B*T, 256]
  __bf16* Vt    = Kb + BT * D_MODEL;          // [B, 256, T]  (transposed V)
  __bf16* ctxb  = Vt + BT * D_MODEL;          // [B*T, 256]
  __bf16* Wqkvt = ctxb + BT * D_MODEL;        // [768, 256]   (W_qkv^T, bf16)
  __bf16* Wpt   = Wqkvt + 3 * D_MODEL * D_MODEL;  // [256, 256] (W_proj^T, bf16)
  // total workspace use: 32 MiB + 512 KiB

  wtrans_kernel<<<dim3((3 * D_MODEL * D_MODEL) / 256), dim3(256), 0, stream>>>(
      Wqkv, Wqkvt, D_MODEL, 3 * D_MODEL);
  wtrans_kernel<<<dim3((D_MODEL * D_MODEL) / 256), dim3(256), 0, stream>>>(
      Wp, Wpt, D_MODEL, D_MODEL);
  qkv_kernel<<<dim3((unsigned)(BT / 64)), dim3(128), 0, stream>>>(x, Wqkvt, bqkv, Qb, Kb, Vt);
  attn_kernel<<<dim3(SEQ_T / 64, N_BATCH), dim3(128), 0, stream>>>(Qb, Kb, Vt, ctxb);
  proj_kernel<<<dim3((unsigned)(BT / 64)), dim3(128), 0, stream>>>(ctxb, Wpt, bp, out);
}
